// AttentionDecoder_59004260712868
// MI455X (gfx1250) — matrix operand order repaired and checked by reading, not compiled, not run on hardware
//
#include <hip/hip_runtime.h>
#include <hip/hip_fp16.h>
#include <math.h>

typedef __attribute__((ext_vector_type(16))) _Float16 v16h;
typedef __attribute__((ext_vector_type(8)))  _Float16 v8h;
typedef __attribute__((ext_vector_type(8)))  float    v8f;

#define SHUF16(lo, hi) \
  __builtin_shufflevector(lo, hi, 0,1,2,3,4,5,6,7,8,9,10,11,12,13,14,15)

#define SCHED_FENCE() __builtin_amdgcn_sched_barrier(0)

// ---------------------------------------------------------------------------
// WMMA operand layouts as loaded by this code, A and B alike (gfx1250 wave32, ld = 512 halves everywhere):
// A (16x32 f16, MxK): lane L -> row L%16; K = 16*(L/16)+p (contiguous 16)
// B (32x16 f16, KxN): lane L -> col L%16; K = 16*(L/16)+p (contiguous 16)
// C/D (16x16 f32):    VGPR v, lane L -> M = v + 8*(L/16), N = L%16
// ---------------------------------------------------------------------------
__device__ __forceinline__ v8f wmma16(v16h a, v16h b, v8f c) {
  return __builtin_amdgcn_wmma_f32_16x16x32_f16(false, a, false, b,
                                                (short)0, c, false, false);
}

__device__ __forceinline__ float sigmoidf_(float x) {
  return 1.0f / (1.0f + expf(-x));
}

// ---------------------------------------------------------------------------
__global__ void ad_cvt_f16(const float* __restrict__ src,
                           _Float16* __restrict__ dst, int n) {
  for (int i = blockIdx.x * blockDim.x + threadIdx.x; i < n;
       i += gridDim.x * blockDim.x)
    dst[i] = (_Float16)src[i];
}

__global__ void ad_init_state(const float* __restrict__ h0,
                              const float* __restrict__ c0,
                              float* __restrict__ hstate,
                              float* __restrict__ cstate,
                              _Float16* __restrict__ hf16, int n) {
  int i = blockIdx.x * blockDim.x + threadIdx.x;
  if (i < n) {
    float h = h0[i];
    hstate[i] = h;
    cstate[i] = c0[i];
    hf16[i] = (_Float16)h;
  }
}

// embedding lookup for all (t,b,h), PAD_ID=0 row forced to zero.
__global__ void ad_embed(const int* __restrict__ tok,
                         const float* __restrict__ emb,
                         _Float16* __restrict__ xall, int n) {
  for (int j = blockIdx.x * blockDim.x + threadIdx.x; j < n;
       j += gridDim.x * blockDim.x) {
    int h  = j & 511;
    int bt = j >> 9;
    int b  = bt & 31;
    int t  = bt >> 5;
    int id = tok[b * 64 + t];
    float v = (id == 0) ? 0.0f : emb[id * 512 + h];
    xall[j] = (_Float16)v;
  }
}

// ---------------------------------------------------------------------------
// K=512 phase for LSTM gates, depth-2 software pipeline with sched fences so
// the machine scheduler cannot sink the prefetch loads onto their consumers.
// ---------------------------------------------------------------------------
__device__ __forceinline__ void lstm_gemm_phase(
    const _Float16* __restrict__ A,   // [32,512] activations f16
    const _Float16* __restrict__ W,   // [2048,512] weights f16
    int m0, int n0, int lane, v8f acc[4]) {
  const int r  = lane & 15;
  const int ga = (lane >> 4) << 4;
  const int gb = (lane >> 4) << 4;
  const _Float16* ap = A + (m0 + r) * 512 + ga;
  const _Float16* bp[4];
#pragma unroll
  for (int g = 0; g < 4; ++g) bp[g] = W + (g * 512 + n0 + r) * 512 + gb;

  v8h al[2], ah[2], bl[2][4], bh[2][4];
#pragma unroll
  for (int sl = 0; sl < 2; ++sl) {           // preload k-steps 0 and 1
    const int kk = sl * 32;
    al[sl] = *(const v8h*)(ap + kk);
    ah[sl] = *(const v8h*)(ap + kk + 8);
#pragma unroll
    for (int g = 0; g < 4; ++g) {
      bl[sl][g] = *(const v8h*)(bp[g] + kk);
      bh[sl][g] = *(const v8h*)(bp[g] + kk + 8);
    }
  }
  SCHED_FENCE();

#pragma unroll
  for (int s = 0; s < 16; ++s) {
    const int cur = s & 1;
    v16h aa = SHUF16(al[cur], ah[cur]);
#pragma unroll
    for (int g = 0; g < 4; ++g)
      acc[g] = wmma16(aa, SHUF16(bl[cur][g], bh[cur][g]), acc[g]);
    SCHED_FENCE();
    if (s + 2 < 16) {                        // refill slot two steps ahead
      const int kk = (s + 2) * 32;
      al[cur] = *(const v8h*)(ap + kk);
      ah[cur] = *(const v8h*)(ap + kk + 8);
#pragma unroll
      for (int g = 0; g < 4; ++g) {
        bl[cur][g] = *(const v8h*)(bp[g] + kk);
        bh[cur][g] = *(const v8h*)(bp[g] + kk + 8);
      }
    }
    SCHED_FENCE();
  }
}

// ---------------------------------------------------------------------------
// one LSTM layer step: fused WMMA gates + cell update + residual.
// grid = 64 blocks (2 M-tiles x 32 N-tiles), block = 1 wave.
// ---------------------------------------------------------------------------
__global__ __launch_bounds__(32)
void ad_lstm_step(const _Float16* __restrict__ xin,
                  const _Float16* __restrict__ hin,
                  const _Float16* __restrict__ Wih,
                  const _Float16* __restrict__ Whh,
                  const float*    __restrict__ bih,
                  const float*    __restrict__ bhh,
                  float*          __restrict__ hstate,
                  float*          __restrict__ cstate,
                  _Float16*       __restrict__ hout,
                  _Float16*       __restrict__ outf16,
                  float*          __restrict__ outf32) {
  const int lane = threadIdx.x;
  const int m0 = (blockIdx.x >> 5) * 16;
  const int n0 = (blockIdx.x & 31) * 16;
  const int r  = lane & 15;

  v8f acc[4] = {v8f{}, v8f{}, v8f{}, v8f{}};
  lstm_gemm_phase(xin, Wih, m0, n0, lane, acc);   // x @ Wih^T
  lstm_gemm_phase(hin, Whh, m0, n0, lane, acc);   // h @ Whh^T

  const int col = n0 + r;
  const int rg  = (lane >> 4) * 8;
  float bias[4];
#pragma unroll
  for (int g = 0; g < 4; ++g) {
    int gi = g * 512 + col;
    bias[g] = bih[gi] + bhh[gi];
  }
#pragma unroll
  for (int v = 0; v < 8; ++v) {
    int b   = m0 + rg + v;
    int idx = b * 512 + col;
    float iv = sigmoidf_(acc[0][v] + bias[0]);
    float fv = sigmoidf_(acc[1][v] + bias[1]);
    float gv = tanhf   (acc[2][v] + bias[2]);
    float ov = sigmoidf_(acc[3][v] + bias[3]);
    float cold = cstate[idx];
    float cnew = fv * cold + iv * gv;
    float hnew = ov * tanhf(cnew);
    cstate[idx] = cnew;
    hstate[idx] = hnew;
    hout[idx]   = (_Float16)hnew;
    float resid = hnew + (float)xin[idx];
    if (outf16) outf16[idx] = (_Float16)resid;
    if (outf32) outf32[idx] = resid;
  }
}

// ---------------------------------------------------------------------------
// batched attention over all (b,t) with faithful log-softmax weighting.
// ---------------------------------------------------------------------------
__global__ __launch_bounds__(256)
void ad_attention(const float* __restrict__ context,
                  const float* __restrict__ finals,
                  _Float16*    __restrict__ outs16) {
  __shared__ float sf[512];
  __shared__ float ssc[128];
  __shared__ float slogz;

  const int b = blockIdx.x >> 6;
  const int t = blockIdx.x & 63;
  const int tid = threadIdx.x;
  const float* fin = finals + (t * 32 + b) * 512;
  const float* cb  = context + b * 128 * 512;

  sf[tid]       = fin[tid];
  sf[tid + 256] = fin[tid + 256];
  __syncthreads();

  const int w = tid >> 5, lane = tid & 31;
  for (int s = w; s < 128; s += 8) {
    const float* cr = cb + s * 512;
    float p = 0.0f;
    for (int h = lane; h < 512; h += 32) p += cr[h] * sf[h];
#pragma unroll
    for (int off = 16; off > 0; off >>= 1) p += __shfl_xor(p, off, 32);
    if (lane == 0) ssc[s] = p;
  }
  __syncthreads();

  if (tid < 32) {
    float m = -INFINITY;
    for (int s = lane; s < 128; s += 32) m = fmaxf(m, ssc[s]);
#pragma unroll
    for (int off = 16; off > 0; off >>= 1) m = fmaxf(m, __shfl_xor(m, off, 32));
    float sum = 0.0f;
    for (int s = lane; s < 128; s += 32) sum += expf(ssc[s] - m);
#pragma unroll
    for (int off = 16; off > 0; off >>= 1) sum += __shfl_xor(sum, off, 32);
    if (lane == 0) slogz = m + logf(sum);
  }
  __syncthreads();
  float logz = slogz;
  if (tid < 128) ssc[tid] -= logz;
  __syncthreads();

  for (int h = tid; h < 512; h += 256) {
    float acc = 0.0f;
    for (int s = 0; s < 128; ++s) acc += cb[s * 512 + h] * ssc[s];
    outs16[(b * 64 + t) * 512 + h] = (_Float16)acc;
  }
}

// ---------------------------------------------------------------------------
// logits = outs @ Wc^T + bc.   M=2048, N=32000, K=512.
// A panel 64x512 (64 KB) staged in LDS via ASYNC global->LDS DMA;
// each wave: 4 M-tiles x 1 N-tile; B stream depth-2 pipelined.
// grid = (2048/64) x (32000/128) = 8000 blocks.
// ---------------------------------------------------------------------------
__global__ __launch_bounds__(256)
void ad_logits(const _Float16* __restrict__ A,
               const _Float16* __restrict__ Bw,
               const float*    __restrict__ bc,
               float*          __restrict__ out) {
  __shared__ _Float16 sA[64 * 512];                // 64 KB

  const int tid  = threadIdx.x;
  const int mblk = blockIdx.x / 250;
  const int nblk = blockIdx.x % 250;
  const int m0   = mblk * 64;

  // ---- stage A[m0..m0+63][0..511] into LDS with async DMA ----
  {
    const _Float16* ablk = A + m0 * 512;
    unsigned sbase = (unsigned)(unsigned long long)(uintptr_t)sA;
#pragma unroll
    for (int i = 0; i < 16; ++i) {                 // 4096 x 16B chunks total
      int c = tid + i * 256;
      const _Float16* gp = ablk + c * 8;
      unsigned lp = sbase + c * 16;
      asm volatile("global_load_async_to_lds_b128 %0, %1, off"
                   :: "v"(lp), "v"(gp) : "memory");
    }
    asm volatile("s_wait_asynccnt 0" ::: "memory");
  }
  __syncthreads();

  const int wv   = tid >> 5;
  const int lane = tid & 31;
  const int r    = lane & 15;
  const int ga   = (lane >> 4) << 4;
  const int gb   = (lane >> 4) << 4;
  const int ncol0 = nblk * 128 + wv * 16;

  const _Float16* bp = Bw + (ncol0 + r) * 512 + gb;
  const _Float16* ap = sA + r * 512 + ga;

  v8f acc[4] = {v8f{}, v8f{}, v8f{}, v8f{}};
  v8h blv[2], bhv[2];
#pragma unroll
  for (int sl = 0; sl < 2; ++sl) {
    blv[sl] = *(const v8h*)(bp + sl * 32);
    bhv[sl] = *(const v8h*)(bp + sl * 32 + 8);
  }
  SCHED_FENCE();

#pragma unroll
  for (int s = 0; s < 16; ++s) {
    const int cur = s & 1;
    v16h bb = SHUF16(blv[cur], bhv[cur]);
    const int kc = s * 32;
#pragma unroll
    for (int mi = 0; mi < 4; ++mi) {               // A operand from LDS
      const _Float16* app = ap + mi * 16 * 512 + kc;
      v16h aa = SHUF16(*(const v8h*)app, *(const v8h*)(app + 8));
      acc[mi] = wmma16(aa, bb, acc[mi]);
    }
    SCHED_FENCE();
    if (s + 2 < 16) {                              // refill B two steps ahead
      const int kk = (s + 2) * 32;
      blv[cur] = *(const v8h*)(bp + kk);
      bhv[cur] = *(const v8h*)(bp + kk + 8);
    }
    SCHED_FENCE();
  }

  const int col = ncol0 + r;
  const float bias = bc[col];
  const int rg = (lane >> 4) * 8;
#pragma unroll
  for (int mi = 0; mi < 4; ++mi)
#pragma unroll
    for (int v = 0; v < 8; ++v) {
      int m = m0 + mi * 16 + rg + v;
      out[m * 32000 + col] = acc[mi][v] + bias;
    }
}

__global__ void ad_finalize(const float* __restrict__ hstate,
                            const float* __restrict__ cstate,
                            float* __restrict__ tail) {
  int j = blockIdx.x * blockDim.x + threadIdx.x;
  if (j < 32768)      tail[j] = hstate[j];
  else if (j < 65536) tail[j] = cstate[j - 32768];
}

// ---------------------------------------------------------------------------
extern "C" void kernel_launch(void* const* d_in, const int* in_sizes, int n_in,
                              void* d_out, int out_size, void* d_ws, size_t ws_size,
                              hipStream_t stream) {
  const float* context = (const float*)d_in[0];
  const int*   dec_inp = (const int*)  d_in[1];
  const float* h0      = (const float*)d_in[2];
  const float* c0      = (const float*)d_in[3];
  const float* emb     = (const float*)d_in[4];
  const float* W_ih    = (const float*)d_in[5];
  const float* W_hh    = (const float*)d_in[6];
  const float* b_ih    = (const float*)d_in[7];
  const float* b_hh    = (const float*)d_in[8];
  const float* Wc      = (const float*)d_in[9];
  const float* bc      = (const float*)d_in[10];
  float* out = (float*)d_out;

  char* w = (char*)d_ws;
  _Float16* Wih16  = (_Float16*)w; w += 2u * 2048 * 512 * 2;
  _Float16* Whh16  = (_Float16*)w; w += 2u * 2048 * 512 * 2;
  _Float16* Wc16   = (_Float16*)w; w += 32000u * 512 * 2;
  _Float16* xall   = (_Float16*)w; w += 64u * 32 * 512 * 2;
  _Float16* inp2   = (_Float16*)w; w += 32u * 512 * 2;
  _Float16* hA     = (_Float16*)w; w += 2u * 32 * 512 * 2;
  _Float16* hB     = (_Float16*)w; w += 2u * 32 * 512 * 2;
  float*    hstate = (float*)w;    w += 2u * 32 * 512 * 4;
  float*    cstate = (float*)w;    w += 2u * 32 * 512 * 4;
  float*    finals = (float*)w;    w += 64u * 32 * 512 * 4;
  _Float16* outs16 = (_Float16*)w; w += 2048u * 512 * 2;

  const int LW = 2048 * 512;
  const int LH = 32 * 512;

  ad_cvt_f16<<<4096, 256, 0, stream>>>(W_ih, Wih16, 2 * LW);
  ad_cvt_f16<<<4096, 256, 0, stream>>>(W_hh, Whh16, 2 * LW);
  ad_cvt_f16<<<4096, 256, 0, stream>>>(Wc,   Wc16, 32000 * 512);

  ad_init_state<<<128, 256, 0, stream>>>(h0, c0, hstate, cstate, hA, 2 * LH);
  ad_embed<<<4096, 256, 0, stream>>>(dec_inp, emb, xall, 64 * LH);

  for (int t = 0; t < 64; ++t) {
    const _Float16* hin = (t & 1) ? hB : hA;
    _Float16*       ho  = (t & 1) ? hA : hB;
    ad_lstm_step<<<64, 32, 0, stream>>>(
        xall + t * LH, hin, Wih16, Whh16, b_ih, b_hh,
        hstate, cstate, ho, inp2, nullptr);
    ad_lstm_step<<<64, 32, 0, stream>>>(
        inp2, hin + LH, Wih16 + LW, Whh16 + LW, b_ih + 2048, b_hh + 2048,
        hstate + LH, cstate + LH, ho + LH, nullptr, finals + t * LH);
  }

  ad_attention<<<2048, 256, 0, stream>>>(context, finals, outs16);
  ad_logits<<<8000, 256, 0, stream>>>(outs16, Wc16, bc, out);
  ad_finalize<<<256, 256, 0, stream>>>(hstate, cstate, out + 65536000);
}
